// BinarizeConv2dSDP_89386859364412
// MI455X (gfx1250) — 8-bit matrix operand order repaired and checked by reading, compiled, not run on hardware
//
#include <hip/hip_runtime.h>

typedef __attribute__((ext_vector_type(4)))  int   v4i;
typedef __attribute__((ext_vector_type(8)))  int   v8i;
typedef __attribute__((ext_vector_type(16))) int   v16i;
typedef __attribute__((ext_vector_type(8)))  float v8f;

#define N_    32
#define CIN   256
#define COUT  256
#define H_    56
#define W_    56
#define KTOT  2304
#define KB_N  18
#define XB_BYTES ((size_t)N_*H_*W_*CIN)
#define WP_BYTES ((size_t)COUT*KTOT)

__device__ __forceinline__ signed char sign_fp8(float v) {
    return (signed char)(v > 0.f ? 0x38 : (v < 0.f ? 0xB8 : 0x00));
}

__device__ __forceinline__ int ci_perm(int ci) {
    return (ci & 0x80) | ((ci & 0x10) << 2) | ((ci & 0x60) >> 1) | (ci & 0x0F);
}

__global__ __launch_bounds__(256) void pack_x_kernel(
    const float* __restrict__ x, signed char* __restrict__ xb) {
    __shared__ __align__(16) signed char tile[W_ * CIN];
    const int nh = blockIdx.x;
    const int n = nh / H_, h = nh % H_;
    const float* xp = x + ((size_t)n * CIN * H_ * W_ + (size_t)h * W_);
    const int t = threadIdx.x;
    #pragma unroll 4
    for (int i = 0; i < (W_ * CIN) / 256; ++i) {
        int idx = i * 256 + t;
        int ci = idx / W_, w = idx % W_;
        tile[w * CIN + ci_perm(ci)] = sign_fp8(xp[(size_t)ci * (H_ * W_) + w]);
    }
    __syncthreads();
    int* __restrict__ dst = (int*)(xb + (size_t)nh * (W_ * CIN));
    const int* src = (const int*)tile;
    #pragma unroll
    for (int j = 0; j < (W_ * CIN / 4) / 256; ++j)
        dst[j * 256 + t] = src[j * 256 + t];
}

__global__ __launch_bounds__(256) void pack_w_kernel(
    const float* __restrict__ M, unsigned int* __restrict__ wp) {
    int d = blockIdx.x * 256 + threadIdx.x;
    if (d >= (int)(WP_BYTES / 4)) return;
    int frag = d >> 9;
    int r    = d & 511;
    int lane = r >> 4, v = r & 15;
    int m = lane & 15, hh = lane >> 4;
    int cob = frag / KB_N, kb = frag % KB_N;
    int co  = cob * 16 + m;
    int v8_ = v & 7;
    int Kbase = kb * 128 + ((v >> 3) << 6)
              + ((v8_ & 4) << 3) + ((v8_ & 2) << 2) + (hh << 4) + ((v8_ & 1) << 2);
    unsigned int packed = 0;
    #pragma unroll
    for (int b = 0; b < 4; ++b) {
        int K   = Kbase + b;
        int tap = K >> 8;
        int ci  = K & 255;
        float f = M[((size_t)co * CIN + ci) * 9 + tap];
        packed |= ((unsigned int)(unsigned char)sign_fp8(f)) << (8 * b);
    }
    wp[d] = packed;
}

#define LCOLS 58
#define LDSX  272

__global__ __launch_bounds__(256) void bconv_wmma_kernel(
    const signed char* __restrict__ xb, const signed char* __restrict__ wp,
    const float* __restrict__ alpha, float* __restrict__ out) {
    __shared__ __align__(16) signed char smem[4 * LCOLS * LDSX];
    const int h0 = blockIdx.x * 2;
    const int n  = blockIdx.y;
    const int t  = threadIdx.x;

    for (int c = t; c < 4 * LCOLS * 16; c += 256) {
        int cell = c >> 4, part = c & 15;
        int rr = cell / LCOLS, xcol = cell % LCOLS;
        int y = h0 + rr - 1, xx = xcol - 1;
        v4i val = {0, 0, 0, 0};
        if ((unsigned)y < H_ && (unsigned)xx < W_)
            val = *(const v4i*)(xb + (((size_t)(n * H_ + y) * W_ + xx) << 8)
                                   + (part << 4));
        *(v4i*)(smem + (rr * LCOLS + xcol) * LDSX + (part << 4)) = val;
    }
    __syncthreads();

    const int wid = t >> 5, lane = t & 31;
    const int ln = lane & 15, hh = lane >> 4;
    const int cob0 = wid * 2;
    const int cobase = cob0 << 4;

    int wv[7], rv[7];
    const signed char* pixPtr[7];
    #pragma unroll
    for (int f = 0; f < 7; ++f) {
        int p = f * 16 + ln;
        int r = (p >= W_) ? 1 : 0;
        rv[f] = r; wv[f] = p - r * W_;
        pixPtr[f] = smem + (r * LCOLS + wv[f]) * LDSX + (hh << 6);
    }

    const signed char* pa0 = wp + ((size_t)(cob0 * KB_N) << 11) + (lane << 6);
    const signed char* pa1 = pa0 + ((size_t)KB_N << 11);

    v8f acc0[7], acc1[7];
    #pragma unroll
    for (int f = 0; f < 7; ++f) {
        acc0[f] = (v8f){0,0,0,0,0,0,0,0};
        acc1[f] = (v8f){0,0,0,0,0,0,0,0};
    }

    #pragma unroll
    for (int tap = 0; tap < 9; ++tap) {
        const int kh = tap / 3, kw = tap % 3;
        const int tapoff = (kh * LCOLS + kw) * LDSX;
        #pragma unroll
        for (int cib = 0; cib < 2; ++cib) {
            const int kb = tap * 2 + cib;
            v16i a0 = *(const v16i*)(pa0 + ((size_t)kb << 11));
            v16i a1 = *(const v16i*)(pa1 + ((size_t)kb << 11));
            #pragma unroll
            for (int f = 0; f < 7; ++f) {
                const v4i* bp = (const v4i*)(pixPtr[f] + tapoff + (cib << 7));
                v4i q0 = bp[0], q1 = bp[1], q2 = bp[2], q3 = bp[3];
                v16i b = __builtin_shufflevector(
                    __builtin_shufflevector(q0, q1, 0,1,2,3,4,5,6,7),
                    __builtin_shufflevector(q2, q3, 0,1,2,3,4,5,6,7),
                    0,1,2,3,4,5,6,7,8,9,10,11,12,13,14,15);
                acc0[f] = __builtin_amdgcn_wmma_f32_16x16x128_fp8_fp8(
                    a0, b, (short)0, acc0[f], false, false);
                acc1[f] = __builtin_amdgcn_wmma_f32_16x16x128_fp8_fp8(
                    a1, b, (short)0, acc1[f], false, false);
            }
        }
    }

    float al0[8], al1[8];
    #pragma unroll
    for (int g = 0; g < 8; ++g) {
        al0[g] = alpha[cobase + hh * 8 + g];
        al1[g] = alpha[cobase + 16 + hh * 8 + g];
    }
    #pragma unroll
    for (int f = 0; f < 7; ++f) {
        int y = h0 + rv[f];
        int base0 = ((n * COUT + cobase + hh * 8) * H_ + y) * W_ + wv[f];
        int base1 = base0 + 16 * (H_ * W_);
        #pragma unroll
        for (int g = 0; g < 8; ++g) {
            out[base0 + g * (H_ * W_)] = acc0[f][g] * al0[g];
            out[base1 + g * (H_ * W_)] = acc1[f][g] * al1[g];
        }
    }
}

extern "C" void kernel_launch(void* const* d_in, const int* in_sizes, int n_in,
                              void* d_out, int out_size, void* d_ws, size_t ws_size,
                              hipStream_t stream) {
    const float* x     = (const float*)d_in[0];
    const float* M     = (const float*)d_in[1];
    const float* alpha = (const float*)d_in[2];
    float* out = (float*)d_out;

    signed char* xb = (signed char*)d_ws;
    signed char* wp = (signed char*)d_ws + XB_BYTES;

    pack_x_kernel<<<N_ * H_, 256, 0, stream>>>(x, xb);
    pack_w_kernel<<<(int)((WP_BYTES / 4 + 255) / 256), 256, 0, stream>>>(
        M, (unsigned int*)wp);

    dim3 grid(H_ / 2, N_);
    bconv_wmma_kernel<<<grid, 256, 0, stream>>>(xb, wp, alpha, out);
}
